// PyGAttention_22093311771376
// MI455X (gfx1250) — matrix operand order repaired and checked by reading, not compiled, not run on hardware
//
#include <hip/hip_runtime.h>

#define NUM_MOLS   16384
#define TOTAL_KEYS 1048576
#define DIM        128

typedef __attribute__((ext_vector_type(16))) __bf16 v16bf;
typedef __attribute__((ext_vector_type(8)))  float  v8f;

// ---- monotone f32 <-> u32 mapping for atomic max over signed floats ----
__device__ __forceinline__ unsigned flip_f32(float f) {
    unsigned u = __float_as_uint(f);
    return (u & 0x80000000u) ? ~u : (u | 0x80000000u);
}
__device__ __forceinline__ float unflip_f32(unsigned u) {
    return __uint_as_float((u & 0x80000000u) ? (u & 0x7fffffffu) : ~u);
}

// ---- 0: zero segment accumulators (d_ws is poisoned; must init every call) ----
__global__ void init_seg(unsigned* __restrict__ segmax, float* __restrict__ segsum) {
    int i = blockIdx.x * blockDim.x + threadIdx.x;
    if (i < NUM_MOLS) { segmax[i] = 0u; segsum[i] = 0.0f; }
}

// ---- 1: Wt = (WQ^T @ WK)^T stored N-major: Wt[e*DIM + d] = sum_h WQ[h,d]*WK[h,e]
// N-major layout makes each WMMA lane's 16 K-values contiguous (32B vector load).
__global__ void fold_weights(const float* __restrict__ WQ,
                             const float* __restrict__ WK,
                             __bf16* __restrict__ Wt) {
    int i = blockIdx.x * blockDim.x + threadIdx.x;   // i = d*128 + e
    if (i >= DIM * DIM) return;
    int d = i >> 7, e = i & 127;
    float s = 0.0f;
    for (int h = 0; h < DIM; ++h)
        s = fmaf(WQ[h * DIM + d], WK[h * DIM + e], s);
    Wt[e * DIM + d] = (__bf16)s;                      // transposed store
}

// ---- 2: M = query @ W  via v_wmma_f32_16x16x32_bf16 ----
// One wave -> one 16x16 tile of M; K=128 in four x32 steps.
// A (16x32 bf16), contiguous k order used here: lane<16 holds row=lane, K 0..15; lane>=16 row=lane-16, K 16..31
// B (32x16 bf16), same k order as A: lane<16 holds col=lane, K 0..15; lane>=16 col=lane-16, K 16..31
//   -> with Wt N-major, that's v16bf at Wt + col*DIM + kc + half*16 (contiguous)
// C/D (16x16 f32): VGPR j: lanes0-15 -> M=j, lanes16-31 -> M=8+j; N=lane&15
__global__ __launch_bounds__(256) void proj_m(const float* __restrict__ Q,
                                              const __bf16* __restrict__ Wt,
                                              float* __restrict__ Mout) {
    const int lane = threadIdx.x & 31;
    const int wv   = threadIdx.x >> 5;       // col tile 0..7
    const int half = lane >> 4;              // lane half select
    const int l15  = lane & 15;
    const int row  = blockIdx.x * 16 + l15;  // A row owned by this lane
    const int col  = wv * 16 + l15;          // B col owned by this lane

    v8f acc = {};
    for (int kc = 0; kc < DIM; kc += 32) {
        v16bf a;
        const float* qrow = Q + (size_t)row * DIM + kc + half * 16;
        #pragma unroll
        for (int i = 0; i < 8; ++i) a[i]     = (__bf16)qrow[i];        // K = kc+half*16+i
        #pragma unroll
        for (int i = 0; i < 8; ++i) a[8 + i] = (__bf16)qrow[8 + i];    // K = kc+half*16+8+i
        // contiguous 32B vector load of this lane's 16 B-matrix K-values
        v16bf b = *(const v16bf*)(Wt + (size_t)col * DIM + kc + half * 16);
        acc = __builtin_amdgcn_wmma_f32_16x16x32_bf16(
                  false, a, false, b, (short)0, acc, false, false);
    }
    float* orow = Mout + (size_t)(blockIdx.x * 16 + half * 8) * DIM + col;
    #pragma unroll
    for (int j = 0; j < 8; ++j) orow[(size_t)j * DIM] = acc[j];
}

// ---- 3: streaming scores: score[k] = <key[k], M[idx[k]]> / sqrt(128) ----
__global__ __launch_bounds__(256) void scores_pass(const float* __restrict__ Kb,
                                                   const float* __restrict__ Mbuf,
                                                   const int* __restrict__ idx,
                                                   float* __restrict__ out_scores,
                                                   unsigned* __restrict__ segmax) {
    int k = blockIdx.x * blockDim.x + threadIdx.x;
    if (k >= TOTAL_KEYS) return;
    int id = idx[k];
    const float4* kr = (const float4*)(Kb   + (size_t)k  * DIM);
    const float4* mr = (const float4*)(Mbuf + (size_t)id * DIM);
    float s = 0.0f;
    #pragma unroll 4
    for (int i = 0; i < DIM / 4; ++i) {
        float4 a = kr[i], b = mr[i];
        s = fmaf(a.x, b.x, s); s = fmaf(a.y, b.y, s);
        s = fmaf(a.z, b.z, s); s = fmaf(a.w, b.w, s);
    }
    s *= 0.08838834764831845f;   // 1/sqrt(128)
    out_scores[k] = s;
    atomicMax(&segmax[id], flip_f32(s));
}

// ---- 4: ex = exp(score - segmax); stash ex in weights slot; segment sum ----
__global__ __launch_bounds__(256) void exp_pass(const float* __restrict__ scores,
                                                const int* __restrict__ idx,
                                                const unsigned* __restrict__ segmax,
                                                float* __restrict__ ex_out,
                                                float* __restrict__ segsum) {
    int k = blockIdx.x * blockDim.x + threadIdx.x;
    if (k >= TOTAL_KEYS) return;
    int id = idx[k];
    float e = __expf(scores[k] - unflip_f32(segmax[id]));
    ex_out[k] = e;
    atomicAdd(&segsum[id], e);
}

// ---- 5: weights = ex / segsum[idx] (in place in d_out weights slot) ----
__global__ __launch_bounds__(256) void norm_pass(const int* __restrict__ idx,
                                                 const float* __restrict__ segsum,
                                                 float* __restrict__ weights) {
    int k = blockIdx.x * blockDim.x + threadIdx.x;
    if (k >= TOTAL_KEYS) return;
    weights[k] = weights[k] / segsum[idx[k]];
}

extern "C" void kernel_launch(void* const* d_in, const int* in_sizes, int n_in,
                              void* d_out, int out_size, void* d_ws, size_t ws_size,
                              hipStream_t stream) {
    const float* Q   = (const float*)d_in[0];   // [16384,128]
    const float* Kb  = (const float*)d_in[1];   // [1048576,128]
    const float* WQ  = (const float*)d_in[2];   // [128,128]
    const float* WK  = (const float*)d_in[3];   // [128,128]
    const int*   idx = (const int*)  d_in[4];   // [1048576]

    float* out     = (float*)d_out;
    float* weights = out;                // [TOTAL_KEYS]
    float* scores  = out + TOTAL_KEYS;   // [TOTAL_KEYS]

    // workspace: Wt bf16 (32KB) | M f32 (8MB) | segmax (64KB) | segsum (64KB)
    char* ws = (char*)d_ws;
    __bf16*   Wt     = (__bf16*)ws;
    float*    Mbuf   = (float*)(ws + 32768);
    unsigned* segmax = (unsigned*)(ws + 32768 + (size_t)NUM_MOLS * DIM * 4);
    float*    segsum = (float*)((char*)segmax + (size_t)NUM_MOLS * 4);

    init_seg    <<<NUM_MOLS / 256, 256, 0, stream>>>(segmax, segsum);
    fold_weights<<<(DIM * DIM) / 256, 256, 0, stream>>>(WQ, WK, Wt);
    proj_m      <<<NUM_MOLS / 16, 256, 0, stream>>>(Q, Wt, Mbuf);
    scores_pass <<<TOTAL_KEYS / 256, 256, 0, stream>>>(Kb, Mbuf, idx, scores, segmax);
    exp_pass    <<<TOTAL_KEYS / 256, 256, 0, stream>>>(scores, idx, segmax, weights, segsum);
    norm_pass   <<<TOTAL_KEYS / 256, 256, 0, stream>>>(idx, segsum, weights);
}
